// ButterflyLayer_63359357551152
// MI455X (gfx1250) — matrix operand order repaired and checked by reading, not compiled, not run on hardware
//
#include <hip/hip_runtime.h>

typedef __attribute__((ext_vector_type(16))) _Float16 v16h;
typedef __attribute__((ext_vector_type(8)))  float    v8f;
typedef __attribute__((ext_vector_type(4)))  int      i4x;
typedef __attribute__((ext_vector_type(4)))  float    f4x;
typedef __attribute__((ext_vector_type(2)))  _Float16 h2x;

// ---------------------------------------------------------------------------
// Workspace layout (f16 element offsets)
//   BT tiles are 16(n) x 32(K) f16, K = s*16 + c (stacked [W0;W1]).
//   Lane (n, hi) loads 8 halves at n*32 + hi*8 and 8 at n*32 + 16 + hi*8.
// ---------------------------------------------------------------------------
#define WS_IN_OFF   0           // 16x32            = 512
#define WS_FH_OFF   512         // 5*32 tiles * 512 = 81920
#define WS_SH_OFF   82432       // 5*32 tiles * 512 = 81920
#define WS_OUT_OFF  164352      // 16x32            = 512
#define WS_MID_OFF  164864      // 32*32*16*16      = 262144
#define WS_TOTAL    427008

union ABu { v16h h; i4x q[2]; };  // 32 bytes

// LDS activation buffer: 1024 rows x 16 f16 (32B rows), 16B chunks XOR-swizzled
// by (row>>3)&1 to avoid lane-half bank conflicts.
__device__ __forceinline__ unsigned lds_off(int row, int chunk) {
  return (unsigned)(row * 32) + (unsigned)((chunk ^ ((row >> 3) & 1)) << 4);
}
__device__ __forceinline__ i4x ld16(const unsigned char* buf, int row, int chunk) {
  return *reinterpret_cast<const i4x*>(buf + lds_off(row, chunk));
}
__device__ __forceinline__ void st16(unsigned char* buf, int row, int col, float v) {
  _Float16 h = (_Float16)v;
  *reinterpret_cast<_Float16*>(buf + lds_off(row, col >> 3) + (col & 7) * 2) = h;
}

// ---------------------------------------------------------------------------
// Pre-pass: convert all weights to f16, laid out as WMMA B-operand tiles.
// ---------------------------------------------------------------------------
__global__ __launch_bounds__(256) void convert_weights(
    const float* __restrict__ in_filter,   // (16,16)  [c][n]
    const float* __restrict__ fh_filters,  // (5,32,2,16,16)
    const float* __restrict__ sh_filters,  // (5,32,2,16,16)
    const float* __restrict__ out_filter,  // (16,16)  [c][f]
    const float* __restrict__ mid_dense,   // (32,32,16,16)
    _Float16* __restrict__ ws) {
  for (int idx = blockIdx.x * blockDim.x + threadIdx.x; idx < WS_TOTAL;
       idx += gridDim.x * blockDim.x) {
    float v;
    if (idx < WS_FH_OFF) {                       // input dense (K>=16 zero pad)
      int n = idx >> 5, K = idx & 31;
      v = (K < 16) ? in_filter[(K & 15) * 16 + n] : 0.f;
    } else if (idx < WS_SH_OFF) {                // fh butterfly tiles
      int j = idx - WS_FH_OFF;
      int tile = j >> 9, w = j & 511, n = w >> 5, K = w & 31;
      v = fh_filters[(size_t)(tile * 2 + (K >> 4)) * 256 + (K & 15) * 16 + n];
    } else if (idx < WS_OUT_OFF) {               // sh butterfly tiles
      int j = idx - WS_SH_OFF;
      int tile = j >> 9, w = j & 511, n = w >> 5, K = w & 31;
      v = sh_filters[(size_t)(tile * 2 + (K >> 4)) * 256 + (K & 15) * 16 + n];
    } else if (idx < WS_MID_OFF) {               // output dense (K>=16 zero)
      int j = idx - WS_OUT_OFF;
      int n = j >> 5, K = j & 31;
      v = (K < 16) ? out_filter[(K & 15) * 16 + n] : 0.f;
    } else {                                     // mid weights, layout kept
      v = mid_dense[idx - WS_MID_OFF];
    }
    ws[idx] = (_Float16)v;
  }
}

// ---------------------------------------------------------------------------
// Fused butterfly network: one workgroup per batch element b.
// 8 waves x 8 tiles = 64 WMMA tiles per stage (1024 rows / 16).
// ---------------------------------------------------------------------------
__global__ __launch_bounds__(256) void butterfly_fused(
    const float* __restrict__ in_data,     // (B, 16384)
    const float* __restrict__ in_bias,     // (16)
    const float* __restrict__ fh_biases,   // (5,32,16)
    const float* __restrict__ mid_bias,    // (32,32,16)
    const float* __restrict__ sh_biases,   // (5,32,16)
    const _Float16* __restrict__ W,        // workspace (f16)
    float* __restrict__ out) {             // (B, 16384)
  __shared__ __align__(16) unsigned char smemA[32768];
  __shared__ __align__(16) unsigned char smemB[32768];

  const int tid  = threadIdx.x;
  const int lane = tid & 31;
  const int wav  = tid >> 5;
  const int hi   = lane >> 4;   // selects K chunk / lane half
  const int n    = lane & 15;   // output column
  const int trow = lane & 15;   // A-matrix row within tile
  const int b    = blockIdx.x;

  const i4x zero16 = {0, 0, 0, 0};

  // ---- Stage 0: input dense (K=16, zero-padded to 32), relu -> buf A ----
  {
    ABu bm;
    const _Float16* wp = W + WS_IN_OFF + n * 32 + hi * 8;
    bm.q[0] = ((const i4x*)wp)[0];
    bm.q[1] = ((const i4x*)wp)[2];
    const float bia = in_bias[n];
    const float* xb = in_data + (size_t)b * 16384;
    for (int j = 0; j < 8; ++j) {
      const int t0 = (wav * 8 + j) * 16;
      const float* p = xb + (size_t)(t0 + trow) * 16 + hi * 8;
      f4x f0 = ((const f4x*)p)[0];
      f4x f1 = ((const f4x*)p)[1];
      ABu a;
      a.h[0] = (_Float16)f0[0]; a.h[1] = (_Float16)f0[1];
      a.h[2] = (_Float16)f0[2]; a.h[3] = (_Float16)f0[3];
      a.h[4] = (_Float16)f1[0]; a.h[5] = (_Float16)f1[1];
      a.h[6] = (_Float16)f1[2]; a.h[7] = (_Float16)f1[3];
      a.q[1] = zero16;
      v8f c = {};
      c = __builtin_amdgcn_wmma_f32_16x16x32_f16(false, a.h, false, bm.h,
                                                 (short)0, c, false, false);
#pragma unroll
      for (int r = 0; r < 8; ++r) {
        int row = t0 + r + hi * 8;
        float v = c[r] + bia;
        st16(smemA, row, n, fmaxf(v, 0.f));
      }
    }
  }
  __syncthreads();
  int cur = 0;

  // ---- Stages 1..5: first-half butterfly levels (K=32 GEMM per tile) ----
  for (int lvl = 1; lvl <= 5; ++lvl) {
    unsigned char* src = cur ? smemB : smemA;
    unsigned char* dst = cur ? smemA : smemB;
    const int s_   = 10 - lvl;
    const int Tout = 1 << s_;
    int curk = -1;
    ABu bm; bm.q[0] = zero16; bm.q[1] = zero16;
    float bia = 0.f;
    for (int j = 0; j < 8; ++j) {
      const int r0 = (wav * 8 + j) * 16;       // contiguous dst rows
      const int k  = r0 >> s_;                 // butterfly branch
      if (k != curk) {
        const _Float16* wp =
            W + WS_FH_OFF + (size_t)((lvl - 1) * 32 + k) * 512 + n * 32 + hi * 8;
        bm.q[0] = ((const i4x*)wp)[0];
        bm.q[1] = ((const i4x*)wp)[2];
        bia = fh_biases[((lvl - 1) * 32 + k) * 16 + n];
        curk = k;
      }
      const int t  = (r0 & (Tout - 1)) + trow;
      const int er = ((k >> 1) * Tout + t) * 2;   // even source row
      ABu a;
      a.q[0] = ld16(src, er, hi);        // K 0-7 / 8-15  (even row)
      a.q[1] = ld16(src, er + 1, hi);    // K 16-23 / 24-31 (odd row)
      v8f c = {};
      c = __builtin_amdgcn_wmma_f32_16x16x32_f16(false, a.h, false, bm.h,
                                                 (short)0, c, false, false);
#pragma unroll
      for (int r = 0; r < 8; ++r) {
        int row = r0 + r + hi * 8;
        float v = c[r] + bia;
        st16(dst, row, n, fmaxf(v, 0.f));
      }
    }
    __syncthreads();
    cur ^= 1;
  }

  // ---- Mid stage: per-(k,t) dense 16x16 via packed f16 VALU FMA ----
  {
    unsigned char* src = cur ? smemB : smemA;
    unsigned char* dst = cur ? smemA : smemB;
#pragma unroll
    for (int rr = 0; rr < 4; ++rr) {
      const int row = rr * 256 + tid;          // v5 row = k*32 + t
      const int k = row >> 5, t = row & 31;
      ABu va;
      va.q[0] = ld16(src, row, 0);
      va.q[1] = ld16(src, row, 1);
      h2x acc[8];
#pragma unroll
      for (int p = 0; p < 8; ++p) acc[p] = (h2x){(_Float16)0, (_Float16)0};
      const h2x* wr = (const h2x*)(W + WS_MID_OFF + (size_t)row * 256);
#pragma unroll
      for (int c2 = 0; c2 < 16; ++c2) {
        h2x vc = {va.h[c2], va.h[c2]};
        const h2x* wl = wr + c2 * 8;
#pragma unroll
        for (int p = 0; p < 8; ++p) acc[p] += vc * wl[p];  // v_pk_fma_f16
      }
      const float* mb = mid_bias + row * 16;
      // reshuffle into second-half layout: dst row = (t>>1)*64 + 2k + (t&1)
      const int drow = (t >> 1) * 64 + 2 * k + (t & 1);
#pragma unroll
      for (int p = 0; p < 8; ++p) {
        float f0 = (float)acc[p][0] + mb[2 * p];
        float f1 = (float)acc[p][1] + mb[2 * p + 1];
        h2x o = {(_Float16)fmaxf(f0, 0.f), (_Float16)fmaxf(f1, 0.f)};
        *reinterpret_cast<h2x*>(dst + lds_off(drow, (2 * p) >> 3) +
                                ((2 * p) & 7) * 2) = o;
      }
    }
    __syncthreads();
    cur ^= 1;
  }

  // ---- Stages 6..10: second-half butterfly levels ----
  for (int li = 0; li < 5; ++li) {
    unsigned char* src = cur ? smemB : smemA;
    unsigned char* dst = cur ? smemA : smemB;
    const int M     = 16 >> li;
    const int Lin   = 64 << li;
    const int ntile = Lin >> 5;                // tiles per (m, a)
    int curw = -1;
    ABu bm; bm.q[0] = zero16; bm.q[1] = zero16;
    float bia = 0.f;
    for (int j = 0; j < 8; ++j) {
      const int tile = wav * 8 + j;
      const int m    = tile / (2 * ntile);
      const int rem  = tile - m * (2 * ntile);
      const int a_   = rem / ntile;
      const int ti   = rem - a_ * ntile;
      const int widx = 2 * m + a_;
      if (widx != curw) {
        const _Float16* wp =
            W + WS_SH_OFF + (size_t)(li * 32 + widx) * 512 + n * 32 + hi * 8;
        bm.q[0] = ((const i4x*)wp)[0];
        bm.q[1] = ((const i4x*)wp)[2];
        bia = sh_biases[(li * 32 + widx) * 16 + n];
        curw = widx;
      }
      const int t  = ti * 16 + trow;
      const int er = m * Lin + 2 * t;
      ABu a;
      a.q[0] = ld16(src, er, hi);
      a.q[1] = ld16(src, er + 1, hi);
      v8f c = {};
      c = __builtin_amdgcn_wmma_f32_16x16x32_f16(false, a.h, false, bm.h,
                                                 (short)0, c, false, false);
#pragma unroll
      for (int r = 0; r < 8; ++r) {
        const int Mr = r + hi * 8;
        const int tt = ti * 16 + Mr;
        const int drow = (M > 1)
                             ? ((m >> 1) * 2 * Lin + 4 * tt + 2 * a_ + (m & 1))
                             : (2 * tt + a_);
        float v = c[r] + bia;
        st16(dst, drow, n, fmaxf(v, 0.f));
      }
    }
    __syncthreads();
    cur ^= 1;
  }

  // ---- Output dense (K=16 zero-padded), no bias / no relu, f32 out ----
  {
    unsigned char* src = cur ? smemB : smemA;
    ABu bm;
    const _Float16* wp = W + WS_OUT_OFF + n * 32 + hi * 8;
    bm.q[0] = ((const i4x*)wp)[0];
    bm.q[1] = ((const i4x*)wp)[2];
    float* ob = out + (size_t)b * 16384;
    for (int j = 0; j < 8; ++j) {
      const int t0 = (wav * 8 + j) * 16;
      ABu a;
      a.q[0] = ld16(src, t0 + trow, hi);
      a.q[1] = zero16;
      v8f c = {};
      c = __builtin_amdgcn_wmma_f32_16x16x32_f16(false, a.h, false, bm.h,
                                                 (short)0, c, false, false);
#pragma unroll
      for (int r = 0; r < 8; ++r) {
        const int row = t0 + r + hi * 8;
        ob[(size_t)row * 16 + n] = c[r];
      }
    }
  }
}

// ---------------------------------------------------------------------------
extern "C" void kernel_launch(void* const* d_in, const int* in_sizes, int n_in,
                              void* d_out, int out_size, void* d_ws, size_t ws_size,
                              hipStream_t stream) {
  (void)in_sizes; (void)n_in; (void)out_size; (void)ws_size;
  const float* in_data    = (const float*)d_in[0];
  const float* in_filter  = (const float*)d_in[1];
  const float* in_bias    = (const float*)d_in[2];
  const float* fh_filters = (const float*)d_in[3];
  const float* fh_biases  = (const float*)d_in[4];
  const float* mid_dense  = (const float*)d_in[5];
  const float* mid_bias   = (const float*)d_in[6];
  const float* sh_filters = (const float*)d_in[7];
  const float* sh_biases  = (const float*)d_in[8];
  const float* out_filter = (const float*)d_in[9];
  _Float16* ws = (_Float16*)d_ws;

  convert_weights<<<(WS_TOTAL + 255) / 256, 256, 0, stream>>>(
      in_filter, fh_filters, sh_filters, out_filter, mid_dense, ws);

  butterfly_fused<<<1024, 256, 0, stream>>>(
      in_data, in_bias, fh_biases, mid_bias, sh_biases, ws, (float*)d_out);
}
